// SAGEConv_128849019138
// MI455X (gfx1250) — matrix operand order repaired and checked by reading, not compiled, not run on hardware
//
#include <hip/hip_runtime.h>
#include <hip/hip_bf16.h>

#define N_SRC  100000
#define N_DST  100000
#define NUM_E  1600000
#define IN_F   128
#define POOL_F 64
#define OUT_F  128
#define CAT_F  (IN_F + POOL_F)   // 192

typedef __attribute__((ext_vector_type(16))) __bf16 v16bf;
typedef __attribute__((ext_vector_type(8)))  __bf16 v8bf;
typedef __attribute__((ext_vector_type(8)))  float  v8f;

union BFrag { v16bf v; struct { v8bf lo, hi; } h; };

// Split 8 consecutive f32 into bf16 hi/lo halves of a 16-element fragment.
__device__ __forceinline__ void cvt_split8(float4 a, float4 b,
                                           v16bf& hi, v16bf& lo, int base) {
  float f[8] = {a.x, a.y, a.z, a.w, b.x, b.y, b.z, b.w};
#pragma unroll
  for (int i = 0; i < 8; ++i) {
    __bf16 hv = (__bf16)f[i];
    hi[base + i] = hv;
    lo[base + i] = (__bf16)(f[i] - (float)hv);
  }
}

// Near-fp32 GEMM step: A*B ~= Ahi*Bhi + Ahi*Blo + Alo*Bhi  (f32 accumulate)
__device__ __forceinline__ v8f wmma_bf16x3(v16bf ah, v16bf al,
                                           v16bf bh, v16bf bl, v8f acc) {
  acc = __builtin_amdgcn_wmma_f32_16x16x32_bf16(false, ah, false, bh,
                                                (short)0, acc, false, false);
  acc = __builtin_amdgcn_wmma_f32_16x16x32_bf16(false, ah, false, bl,
                                                (short)0, acc, false, false);
  acc = __builtin_amdgcn_wmma_f32_16x16x32_bf16(false, al, false, bh,
                                                (short)0, acc, false, false);
  return acc;
}

// ---------------------------------------------------------------------------
// Weight prep: transpose W[k][n] -> Wt[n][k] and split into bf16 hi/lo.
// ---------------------------------------------------------------------------
__global__ void __launch_bounds__(256) prep_weights_kernel(
    const float* __restrict__ Wp, const float* __restrict__ Wn,
    __bf16* __restrict__ Wp_hi, __bf16* __restrict__ Wp_lo,
    __bf16* __restrict__ Wn_hi, __bf16* __restrict__ Wn_lo) {
  int idx = blockIdx.x * blockDim.x + threadIdx.x;
  if (idx < IN_F * POOL_F) {
    int k = idx / POOL_F, n = idx % POOL_F;
    float x = Wp[idx];
    __bf16 hv = (__bf16)x;
    Wp_hi[n * IN_F + k] = hv;
    Wp_lo[n * IN_F + k] = (__bf16)(x - (float)hv);
  } else {
    int j = idx - IN_F * POOL_F;
    if (j < CAT_F * OUT_F) {
      int k = j / OUT_F, n = j % OUT_F;
      float x = Wn[j];
      __bf16 hv = (__bf16)x;
      Wn_hi[n * CAT_F + k] = hv;
      Wn_lo[n * CAT_F + k] = (__bf16)(x - (float)hv);
    }
  }
}

// ---------------------------------------------------------------------------
// GEMM1: hp = relu(h @ W_pool + b_pool).  4 waves/block, wave w -> N tile w*16.
// ---------------------------------------------------------------------------
__global__ void __launch_bounds__(128) gemm_pool_kernel(
    const float* __restrict__ h, const __bf16* __restrict__ Wt_hi,
    const __bf16* __restrict__ Wt_lo, const float* __restrict__ b_pool,
    float* __restrict__ hp) {
  const int wave = threadIdx.x >> 5;
  const int lane = threadIdx.x & 31;
  const int hs   = lane >> 4;      // half-select (K-group)
  const int l16  = lane & 15;
  const int m0   = blockIdx.x * 16;
  const int n    = wave * 16 + l16;

  float bias = b_pool[n];
  v8f acc;
#pragma unroll
  for (int r = 0; r < 8; ++r) acc[r] = bias;

  int mA = m0 + l16; if (mA >= N_SRC) mA = N_SRC - 1;  // clamp (EXEC stays full)
  const float* rowA = h + (size_t)mA * IN_F;

#pragma unroll
  for (int kc = 0; kc < IN_F / 32; ++kc) {
    const int kk = kc * 32;
    v16bf ah, al;
    const float4* pa = (const float4*)(rowA + kk + 8 * hs);
    cvt_split8(pa[0], pa[1], ah, al, 0);
    const float4* pb = (const float4*)(rowA + kk + 8 * hs + 16);
    cvt_split8(pb[0], pb[1], ah, al, 8);

    BFrag bh, bl;
    const v8bf* wh = (const v8bf*)(Wt_hi + n * IN_F + kk + 8 * hs);
    bh.h.lo = wh[0]; bh.h.hi = wh[2];
    const v8bf* wl = (const v8bf*)(Wt_lo + n * IN_F + kk + 8 * hs);
    bl.h.lo = wl[0]; bl.h.hi = wl[2];

    acc = wmma_bf16x3(ah, al, bh.v, bl.v, acc);
  }

#pragma unroll
  for (int r = 0; r < 8; ++r) {
    int m = m0 + r + 8 * hs;
    float v = acc[r];
    v = v > 0.f ? v : 0.f;               // relu
    if (m < N_SRC) hp[(size_t)m * POOL_F + n] = v;
  }
}

// ---------------------------------------------------------------------------
// Degree count: deg[src] += 1  (hardware f32 atomics, L2-resident)
// ---------------------------------------------------------------------------
__global__ void __launch_bounds__(256) count_deg_kernel(
    const int* __restrict__ edge_src, float* __restrict__ deg) {
  int e = blockIdx.x * blockDim.x + threadIdx.x;
  if (e < NUM_E) unsafeAtomicAdd(&deg[edge_src[e]], 1.0f);
}

// Pre-scale hp rows by 1/deg once (hoists division out of the E*64 scatter).
__global__ void __launch_bounds__(256) scale_hp_kernel(
    float* __restrict__ hp, const float* __restrict__ deg) {
  int i = blockIdx.x * blockDim.x + threadIdx.x;
  if (i < N_SRC * POOL_F) {
    float d = deg[i >> 6];               // POOL_F == 64
    if (d > 0.f) hp[i] /= d;
  }
}

// Scatter-add: agg[dst] += hp_scaled[src].  64 lanes per edge, coalesced.
__global__ void __launch_bounds__(256) scatter_kernel(
    const int* __restrict__ edge_src, const int* __restrict__ edge_dst,
    const float* __restrict__ hp, float* __restrict__ agg) {
  int idx = blockIdx.x * blockDim.x + threadIdx.x;
  int e = idx >> 6, f = idx & 63;
  if (e < NUM_E) {
    int s = edge_src[e], d = edge_dst[e];
    unsafeAtomicAdd(&agg[(size_t)d * POOL_F + f], hp[(size_t)s * POOL_F + f]);
  }
}

// ---------------------------------------------------------------------------
// GEMM2: out = concat(h_self, agg) @ W_neigh + b_neigh.  8 waves cover N=128.
// ---------------------------------------------------------------------------
__global__ void __launch_bounds__(256) gemm_out_kernel(
    const float* __restrict__ h_self, const float* __restrict__ agg,
    const __bf16* __restrict__ Wt_hi, const __bf16* __restrict__ Wt_lo,
    const float* __restrict__ b_neigh, float* __restrict__ out) {
  const int wave = threadIdx.x >> 5;
  const int lane = threadIdx.x & 31;
  const int hs   = lane >> 4;
  const int l16  = lane & 15;
  const int m0   = blockIdx.x * 16;
  const int n    = wave * 16 + l16;

  float bias = b_neigh[n];
  v8f acc;
#pragma unroll
  for (int r = 0; r < 8; ++r) acc[r] = bias;

  int mA = m0 + l16; if (mA >= N_DST) mA = N_DST - 1;
  const float* rowH = h_self + (size_t)mA * IN_F;
  const float* rowG = agg    + (size_t)mA * POOL_F;

#pragma unroll
  for (int kc = 0; kc < CAT_F / 32; ++kc) {     // 6 chunks; 0..3 from h_self
    const int kk = kc * 32;
    const float* src = (kk < IN_F) ? (rowH + kk) : (rowG + (kk - IN_F));
    v16bf ah, al;
    const float4* pa = (const float4*)(src + 8 * hs);
    cvt_split8(pa[0], pa[1], ah, al, 0);
    const float4* pb = (const float4*)(src + 8 * hs + 16);
    cvt_split8(pb[0], pb[1], ah, al, 8);

    BFrag bh, bl;
    const v8bf* wh = (const v8bf*)(Wt_hi + n * CAT_F + kk + 8 * hs);
    bh.h.lo = wh[0]; bh.h.hi = wh[2];
    const v8bf* wl = (const v8bf*)(Wt_lo + n * CAT_F + kk + 8 * hs);
    bl.h.lo = wl[0]; bl.h.hi = wl[2];

    acc = wmma_bf16x3(ah, al, bh.v, bl.v, acc);
  }

#pragma unroll
  for (int r = 0; r < 8; ++r) {
    int m = m0 + r + 8 * hs;
    if (m < N_DST) out[(size_t)m * OUT_F + n] = acc[r];
  }
}

extern "C" void kernel_launch(void* const* d_in, const int* in_sizes, int n_in,
                              void* d_out, int out_size, void* d_ws, size_t ws_size,
                              hipStream_t stream) {
  const float* h       = (const float*)d_in[0];
  const float* h_self  = (const float*)d_in[1];
  const int*   edge_dst= (const int*)d_in[2];
  const int*   edge_src= (const int*)d_in[3];
  const float* W_pool  = (const float*)d_in[4];
  const float* b_pool  = (const float*)d_in[5];
  const float* W_neigh = (const float*)d_in[6];
  const float* b_neigh = (const float*)d_in[7];
  float* out = (float*)d_out;

  // Workspace carve-up (all offsets 16B aligned).
  char* ws = (char*)d_ws;
  float* hp  = (float*)ws;  ws += (size_t)N_SRC * POOL_F * sizeof(float);
  float* deg = (float*)ws;  ws += (size_t)N_SRC * sizeof(float);
  float* agg = (float*)ws;  ws += (size_t)N_DST * POOL_F * sizeof(float);
  __bf16* Wp_hi = (__bf16*)ws; ws += (size_t)IN_F  * POOL_F * sizeof(__bf16);
  __bf16* Wp_lo = (__bf16*)ws; ws += (size_t)IN_F  * POOL_F * sizeof(__bf16);
  __bf16* Wn_hi = (__bf16*)ws; ws += (size_t)CAT_F * OUT_F  * sizeof(__bf16);
  __bf16* Wn_lo = (__bf16*)ws; ws += (size_t)CAT_F * OUT_F  * sizeof(__bf16);

  hipMemsetAsync(deg, 0, (size_t)N_SRC * sizeof(float), stream);
  hipMemsetAsync(agg, 0, (size_t)N_DST * POOL_F * sizeof(float), stream);

  int prep_elems = IN_F * POOL_F + CAT_F * OUT_F;
  prep_weights_kernel<<<(prep_elems + 255) / 256, 256, 0, stream>>>(
      W_pool, W_neigh, Wp_hi, Wp_lo, Wn_hi, Wn_lo);

  gemm_pool_kernel<<<(N_SRC + 15) / 16, 128, 0, stream>>>(
      h, Wp_hi, Wp_lo, b_pool, hp);

  count_deg_kernel<<<(NUM_E + 255) / 256, 256, 0, stream>>>(edge_src, deg);

  scale_hp_kernel<<<(N_SRC * POOL_F + 255) / 256, 256, 0, stream>>>(hp, deg);

  scatter_kernel<<<((size_t)NUM_E * 64 + 255) / 256, 256, 0, stream>>>(
      edge_src, edge_dst, hp, agg);

  gemm_out_kernel<<<(N_DST + 15) / 16, 256, 0, stream>>>(
      h_self, agg, Wn_hi, Wn_lo, b_neigh, out);
}
